// SynapForge100M_33852932227185
// MI455X (gfx1250) — matrix operand order repaired and checked by reading, not compiled, not run on hardware
//
#include <hip/hip_runtime.h>
#include <math.h>

// ---------------------------------------------------------------------------
// SynapForge100M forward pass for gfx1250 (MI455X, wave32, WMMA bf16 + TDM).
// B=4 T=256 D=512 L=10 H=4096 V=50257, LOOP=4.
// ---------------------------------------------------------------------------

#define Bb   4
#define Tt   256
#define Dd   512
#define Ll   10
#define Hh   4096
#define Vv   50257
#define LOOPN 4
#define THV  0.05f
#define EPSV 1e-6f
#define ROWS (Bb*Tt)          // 1024
#define KCH  128              // K-panel depth staged via TDM

typedef __attribute__((ext_vector_type(16))) __bf16       v16bf;
typedef __attribute__((ext_vector_type(8)))  float        v8f;
typedef __attribute__((ext_vector_type(4)))  unsigned int u32x4;
typedef __attribute__((ext_vector_type(8)))  unsigned int u32x8;

static_assert(sizeof(v16bf) == 32, "v16bf size");

// ------------------------- helpers -----------------------------------------

__device__ __forceinline__ float sigmoidf_(float x) {
  return 1.0f / (1.0f + __expf(-x));
}

__device__ __forceinline__ v8f vzero8() {
  v8f z;
#pragma unroll
  for (int i = 0; i < 8; ++i) z[i] = 0.0f;
  return z;
}

// A fragment: 16x32 bf16 tile of row-major fp32 A, contiguous k order (k = 16*h + e).
// lanes 0-15: rows 0-15, K{0..15}; lanes 16-31: rows 0-15, K{16..31}.
__device__ __forceinline__ v16bf load_a_frag(const float* __restrict__ A, int lda,
                                             int m0, int k0) {
  const int lane = threadIdx.x & 31;
  const float* p = A + (size_t)(m0 + (lane & 15)) * lda + k0 + ((lane >> 4) << 4);
  v16bf u;
#pragma unroll
  for (int e = 0; e < 8; ++e) {
    u[e]     = (__bf16)p[e];
    u[8 + e] = (__bf16)p[8 + e];
  }
  return u;
}

// B fragment from an LDS-staged fp32 panel laid out [k][64].
// lane: col (L&15); K = kk + 16*(L>=16) + e, e=0..15.
__device__ __forceinline__ v16bf load_b_frag_lds(const float* sm, int kk, int ncol0) {
  const int lane = threadIdx.x & 31;
  const float* p = sm + (kk + ((lane >> 4) << 4)) * 64 + ncol0 + (lane & 15);
  v16bf u;
#pragma unroll
  for (int e = 0; e < 16; ++e) u[e] = (__bf16)p[e * 64];
  return u;
}

// B fragment, transposed source: W is (N,K) row-major; B[k,n] = W[n,k].
__device__ __forceinline__ v16bf load_b_frag_t(const float* __restrict__ W, int ldw,
                                               int k0, int n0) {
  const int lane = threadIdx.x & 31;
  const float* p = W + (size_t)(n0 + (lane & 15)) * ldw + k0 + ((lane >> 4) << 4);
  v16bf u;
#pragma unroll
  for (int e = 0; e < 16; ++e) u[e] = (__bf16)p[e];
  return u;
}

// Transposed + boolean mask (95%-sparse synapse): B[k,n] = W[n,k]*M[n,k].
__device__ __forceinline__ v16bf load_b_frag_tm(const float* __restrict__ W,
                                                const unsigned char* __restrict__ Mk,
                                                int ldw, int k0, int n0) {
  const int lane = threadIdx.x & 31;
  const size_t off = (size_t)(n0 + (lane & 15)) * ldw + k0 + ((lane >> 4) << 4);
  v16bf u;
#pragma unroll
  for (int e = 0; e < 16; ++e)
    u[e] = (__bf16)(Mk[off + e] ? W[off + e] : 0.0f);
  return u;
}

// Transposed with N-bound guard (LM head, V not multiple of 16).
__device__ __forceinline__ v16bf load_b_frag_tg(const float* __restrict__ W, int ldw,
                                                int k0, int n0, int nmax) {
  const int lane = threadIdx.x & 31;
  const int col = n0 + (lane & 15);
  v16bf u;
  if (col < nmax) {
    const float* p = W + (size_t)col * ldw + k0 + ((lane >> 4) << 4);
#pragma unroll
    for (int e = 0; e < 16; ++e) u[e] = (__bf16)p[e];
  } else {
#pragma unroll
    for (int e = 0; e < 16; ++e) u[e] = (__bf16)0.0f;
  }
  return u;
}

__device__ __forceinline__ v8f wmma_bf16(v16bf a, v16bf b, v8f c) {
  return __builtin_amdgcn_wmma_f32_16x16x32_bf16(false, a, false, b, (short)0, c,
                                                 false, false);
}

// C/D layout: lane -> col n0+(L&15); VGPR r -> row m0 + r + 8*(L>=16).
__device__ __forceinline__ void store_tile(float* __restrict__ O, int ldo,
                                           int m0, int n0, v8f acc) {
  const int lane  = threadIdx.x & 31;
  const int col   = n0 + (lane & 15);
  const int rbase = m0 + ((lane >> 4) << 3);
#pragma unroll
  for (int r = 0; r < 8; ++r) O[(size_t)(rbase + r) * ldo + col] = acc[r];
}

// ------------------------- Tensor Data Mover --------------------------------
// Build a 2D D# (ISA cdna5 ch.8) and issue TENSOR_LOAD_TO_LDS via inline asm
// (portable across both toolchains' clang builtins). Copies rows x cols fp32
// elements, row stride = row_stride elements, into LDS at lds_off (bytes),
// packed contiguously ([row][col], col fastest).
__device__ __forceinline__ void tdm_load_2d(unsigned lds_off,
                                            const float* gptr,
                                            unsigned rows, unsigned cols,
                                            unsigned row_stride) {
  const unsigned long long ga = (unsigned long long)(uintptr_t)gptr;
  const unsigned long long s0 = (unsigned long long)row_stride;
  u32x4 g0;
  g0[0] = 1u;                                              // count=1 (valid D#)
  g0[1] = lds_off;                                         // lds_addr bytes
  g0[2] = (unsigned)ga;                                    // global_addr[31:0]
  g0[3] = (unsigned)((ga >> 32) & 0x01FFFFFFull) | (2u << 30);  // [56:32]|type=2
  u32x8 g1;
  g1[0] = (2u << 16);                                      // data_size=4B, mask=0
  g1[1] = (cols & 0xFFFFu) << 16;                          // tensor_dim0[15:0]
  g1[2] = ((cols >> 16) & 0xFFFFu) | ((rows & 0xFFFFu) << 16); // dim0 hi | dim1 lo
  g1[3] = ((rows >> 16) & 0xFFFFu) | ((cols & 0xFFFFu) << 16); // dim1 hi | tile0
  g1[4] = (rows & 0xFFFFu);                                // tile1 | tile2=0
  g1[5] = (unsigned)s0;                                    // dim0_stride[31:0]
  g1[6] = (unsigned)((s0 >> 32) & 0xFFFFu);                // stride0 hi | stride1 lo
  g1[7] = 0u;                                              // stride1 hi
  asm volatile("tensor_load_to_lds %0, %1" :: "s"(g0), "s"(g1) : "memory");
}

// ------------------------- elementwise kernels ------------------------------

__global__ __launch_bounds__(256) void embed_kernel(const int* __restrict__ ids,
                                                    const float* __restrict__ tok,
                                                    const float* __restrict__ pos,
                                                    float* __restrict__ x) {
  const int row = blockIdx.x;            // b*T + t
  const int d   = threadIdx.x;
  const int t   = row & (Tt - 1);
  const int id  = ids[row];
  const float* tr = tok + (size_t)id * Dd;
  const float* pr = pos + (size_t)t * Dd;
  float* xr = x + (size_t)row * Dd;
  xr[d]       = tr[d]       + pr[d];
  xr[d + 256] = tr[d + 256] + pr[d + 256];
}

__global__ __launch_bounds__(256) void rmsnorm_kernel(const float* __restrict__ x,
                                                      const float* __restrict__ w,
                                                      float* __restrict__ out) {
  const int row = blockIdx.x;
  const int tid = threadIdx.x;
  const float* xr = x + (size_t)row * Dd;
  float v0 = xr[tid], v1 = xr[tid + 256];
  float ss = v0 * v0 + v1 * v1;
#pragma unroll
  for (int off = 16; off > 0; off >>= 1) ss += __shfl_xor(ss, off, 32);
  __shared__ float red[8];
  const int wid = tid >> 5, lane = tid & 31;
  if (lane == 0) red[wid] = ss;
  __syncthreads();
  float tot = 0.0f;
#pragma unroll
  for (int i = 0; i < 8; ++i) tot += red[i];
  const float rms = rsqrtf(tot * (1.0f / (float)Dd) + EPSV);
  float* orow = out + (size_t)row * Dd;
  orow[tid]       = v0 * rms * w[tid];
  orow[tid + 256] = v1 * rms * w[tid + 256];
}

// PLIF membrane scan: per-(b,d) channel is independent -> one thread each.
__global__ __launch_bounds__(256) void plif_kernel(const float* __restrict__ h,
                                                   const float* __restrict__ pw,
                                                   float* __restrict__ s) {
  const int idx = blockIdx.x * 256 + threadIdx.x;   // 0..2047
  const int b = idx >> 9, d = idx & (Dd - 1);
  const float decay = sigmoidf_(pw[0]);
  const float* hp = h + (size_t)b * Tt * Dd + d;
  float* sp = s + (size_t)b * Tt * Dd + d;
  float v = 0.0f;
  for (int t = 0; t < Tt; ++t) {
    v += (hp[(size_t)t * Dd] - v) * decay;
    const float spike = (v - THV >= 0.0f) ? 1.0f : 0.0f;
    v -= spike * THV;
    sp[(size_t)t * Dd] = spike;
  }
}

// CfC liquid cell recurrence: one workgroup per batch element, h in LDS.
__global__ __launch_bounds__(512) void liquid_kernel(const float* __restrict__ xpf,
                                                     const float* __restrict__ xpg,
                                                     const float* __restrict__ xph,
                                                     const float* __restrict__ WfL,
                                                     const float* __restrict__ WgL,
                                                     const float* __restrict__ WhL,
                                                     float* __restrict__ hout) {
  const int b = blockIdx.x;
  const int d = threadIdx.x;                 // 0..511
  __shared__ float hs[Dd];
  hs[d] = 0.0f;
  __syncthreads();
  const float* WfH = WfL + (size_t)Dd * Dd;  // h-part rows of (2D,D)
  const float* WgH = WgL + (size_t)Dd * Dd;
  const float* WhH = WhL + (size_t)Dd * Dd;
  const size_t rb = (size_t)b * Tt * Dd;
  for (int t = 0; t < Tt; ++t) {
    const size_t ro = rb + (size_t)t * Dd + d;
    float uf = xpf[ro], ug = xpg[ro], uh = xph[ro];
#pragma unroll 8
    for (int k = 0; k < Dd; ++k) {
      const float hk = hs[k];
      const size_t wo = (size_t)k * Dd + d;
      uf = fmaf(hk, WfH[wo], uf);
      ug = fmaf(hk, WgH[wo], ug);
      uh = fmaf(hk, WhH[wo], uh);
    }
    const float sg = 1.0f / (1.0f + __expf(uf));   // sigmoid(-(z@Wf))
    const float hn = sg * tanhf(ug) + (1.0f - sg) * tanhf(uh);
    __syncthreads();
    hs[d] = hn;
    hout[ro] = hn;
    __syncthreads();
  }
}

// ------------------------- WMMA GEMM kernels --------------------------------
// Block = 128 threads (4 waves); each wave owns a 16x64 output strip.
// "Normal"-layout B panels are staged in LDS by the Tensor Data Mover.

// xproj: of/og/oh = a @ W*[:D,:]  (z-dim selects Wf/Wg/Wh; TDM-staged B)
__global__ __launch_bounds__(128) void gemm_xproj(const float* __restrict__ A,
                                                  const float* __restrict__ Wf,
                                                  const float* __restrict__ Wg,
                                                  const float* __restrict__ Wh,
                                                  float* __restrict__ of,
                                                  float* __restrict__ og,
                                                  float* __restrict__ oh) {
  const float* W = (blockIdx.z == 0) ? Wf : (blockIdx.z == 1 ? Wg : Wh);
  float*       O = (blockIdx.z == 0) ? of : (blockIdx.z == 1 ? og : oh);
  __shared__ float sp[KCH * 64];
  const int wid = threadIdx.x >> 5;
  const int m0 = blockIdx.y * 64 + wid * 16;
  const int n0 = blockIdx.x * 64;
  const unsigned sp_off = (unsigned)(uintptr_t)(void*)sp;
  v8f acc[4];
#pragma unroll
  for (int j = 0; j < 4; ++j) acc[j] = vzero8();
  for (int k0 = 0; k0 < Dd; k0 += KCH) {
    __syncthreads();                               // previous panel fully read
    if (threadIdx.x < 32) {
      tdm_load_2d(sp_off, W + (size_t)k0 * Dd + n0, KCH, 64, Dd);
      __builtin_amdgcn_s_wait_tensorcnt(0);
    }
    __syncthreads();                               // publish panel to all waves
#pragma unroll
    for (int kk = 0; kk < KCH; kk += 32) {
      __builtin_prefetch(A + (size_t)(m0 + (threadIdx.x & 15)) * Dd + k0 + kk + 32, 0, 1);
      const v16bf af = load_a_frag(A, Dd, m0, k0 + kk);
#pragma unroll
      for (int j = 0; j < 4; ++j)
        acc[j] = wmma_bf16(af, load_b_frag_lds(sp, kk, j * 16), acc[j]);
    }
  }
#pragma unroll
  for (int j = 0; j < 4; ++j) store_tile(O, Dd, m0, n0 + j * 16, acc[j]);
}

// x += (s @ (sw*sm)^T) * sigmoid(s @ gw^T + gb)     (fused, in-place residual)
__global__ __launch_bounds__(128) void gemm_syn_gate(const float* __restrict__ S,
                                                     const float* __restrict__ sw,
                                                     const unsigned char* __restrict__ sm,
                                                     const float* __restrict__ gw,
                                                     const float* __restrict__ gb,
                                                     float* __restrict__ X) {
  const int wid = threadIdx.x >> 5;
  const int m0 = blockIdx.y * 64 + wid * 16;
  const int n0 = blockIdx.x * 64;
  v8f a1[4], a2[4];
#pragma unroll
  for (int j = 0; j < 4; ++j) { a1[j] = vzero8(); a2[j] = vzero8(); }
  for (int k0 = 0; k0 < Dd; k0 += 32) {
    const v16bf af = load_a_frag(S, Dd, m0, k0);
#pragma unroll
    for (int j = 0; j < 4; ++j) {
      a1[j] = wmma_bf16(af, load_b_frag_tm(sw, sm, Dd, k0, n0 + j * 16), a1[j]);
      a2[j] = wmma_bf16(af, load_b_frag_t(gw, Dd, k0, n0 + j * 16), a2[j]);
    }
  }
  const int lane  = threadIdx.x & 31;
  const int rbase = m0 + ((lane >> 4) << 3);
#pragma unroll
  for (int j = 0; j < 4; ++j) {
    const int col = n0 + j * 16 + (lane & 15);
    const float bias = gb[col];
#pragma unroll
    for (int r = 0; r < 8; ++r) {
      const size_t o = (size_t)(rbase + r) * Dd + col;
      X[o] += a1[j][r] * sigmoidf_(a2[j][r] + bias);
    }
  }
}

// t = silu(b @ fwg) * (b @ fwu)  -- TDM stages 128x64 fp32 B panels in LDS.
__global__ __launch_bounds__(128) void gemm_ffn1(const float* __restrict__ A,
                                                 const float* __restrict__ Wg,
                                                 const float* __restrict__ Wu,
                                                 float* __restrict__ O) {
  __shared__ float spanel[2 * KCH * 64];           // [0]=gate, [1]=up panels
  float* sg = spanel;
  float* su = spanel + KCH * 64;
  const int wid = threadIdx.x >> 5;
  const int m0 = blockIdx.y * 64 + wid * 16;
  const int n0 = blockIdx.x * 64;
  v8f ag[4], au[4];
#pragma unroll
  for (int j = 0; j < 4; ++j) { ag[j] = vzero8(); au[j] = vzero8(); }
  const unsigned sg_off = (unsigned)(uintptr_t)(void*)sg;
  const unsigned su_off = (unsigned)(uintptr_t)(void*)su;

  for (int k0 = 0; k0 < Dd; k0 += KCH) {
    __syncthreads();                               // previous panel fully read
    if (threadIdx.x < 32) {                        // wave 0 drives the TDM
      tdm_load_2d(sg_off, Wg + (size_t)k0 * Hh + n0, KCH, 64, Hh);
      tdm_load_2d(su_off, Wu + (size_t)k0 * Hh + n0, KCH, 64, Hh);
      __builtin_amdgcn_s_wait_tensorcnt(0);
    }
    __syncthreads();                               // publish panel to all waves
#pragma unroll
    for (int kk = 0; kk < KCH; kk += 32) {
      const v16bf af = load_a_frag(A, Dd, m0, k0 + kk);
#pragma unroll
      for (int j = 0; j < 4; ++j) {
        ag[j] = wmma_bf16(af, load_b_frag_lds(sg, kk, j * 16), ag[j]);
        au[j] = wmma_bf16(af, load_b_frag_lds(su, kk, j * 16), au[j]);
      }
    }
  }
  const int lane  = threadIdx.x & 31;
  const int rbase = m0 + ((lane >> 4) << 3);
#pragma unroll
  for (int j = 0; j < 4; ++j) {
    const int col = n0 + j * 16 + (lane & 15);
#pragma unroll
    for (int r = 0; r < 8; ++r) {
      const float g = ag[j][r];
      O[(size_t)(rbase + r) * Hh + col] = g * sigmoidf_(g) * au[j][r];
    }
  }
}

// x += t @ fwd   (residual accumulate; K=4096, TDM-staged B panels)
__global__ __launch_bounds__(128) void gemm_ffn2(const float* __restrict__ A,
                                                 const float* __restrict__ Wd,
                                                 float* __restrict__ X) {
  __shared__ float sp[KCH * 64];
  const int wid = threadIdx.x >> 5;
  const int m0 = blockIdx.y * 64 + wid * 16;
  const int n0 = blockIdx.x * 64;
  const unsigned sp_off = (unsigned)(uintptr_t)(void*)sp;
  v8f acc[4];
#pragma unroll
  for (int j = 0; j < 4; ++j) acc[j] = vzero8();
  for (int k0 = 0; k0 < Hh; k0 += KCH) {
    __syncthreads();                               // previous panel fully read
    if (threadIdx.x < 32) {
      tdm_load_2d(sp_off, Wd + (size_t)k0 * Dd + n0, KCH, 64, Dd);
      __builtin_amdgcn_s_wait_tensorcnt(0);
    }
    __syncthreads();                               // publish panel to all waves
#pragma unroll
    for (int kk = 0; kk < KCH; kk += 32) {
      __builtin_prefetch(A + (size_t)(m0 + (threadIdx.x & 15)) * Hh + k0 + kk + 32, 0, 1);
      const v16bf af = load_a_frag(A, Hh, m0, k0 + kk);
#pragma unroll
      for (int j = 0; j < 4; ++j)
        acc[j] = wmma_bf16(af, load_b_frag_lds(sp, kk, j * 16), acc[j]);
    }
  }
  const int lane  = threadIdx.x & 31;
  const int rbase = m0 + ((lane >> 4) << 3);
#pragma unroll
  for (int j = 0; j < 4; ++j) {
    const int col = n0 + j * 16 + (lane & 15);
#pragma unroll
    for (int r = 0; r < 8; ++r) X[(size_t)(rbase + r) * Dd + col] += acc[j][r];
  }
}

// logits = xf @ tok_embed^T  (N = V, guarded; B rows contiguous per lane)
__global__ __launch_bounds__(128) void gemm_lmhead(const float* __restrict__ A,
                                                   const float* __restrict__ E,
                                                   float* __restrict__ O) {
  const int wid = threadIdx.x >> 5;
  const int m0 = blockIdx.y * 64 + wid * 16;
  const int n0 = blockIdx.x * 64;
  v8f acc[4];
#pragma unroll
  for (int j = 0; j < 4; ++j) acc[j] = vzero8();
  for (int k0 = 0; k0 < Dd; k0 += 32) {
    const v16bf af = load_a_frag(A, Dd, m0, k0);
#pragma unroll
    for (int j = 0; j < 4; ++j)
      acc[j] = wmma_bf16(af, load_b_frag_tg(E, Dd, k0, n0 + j * 16, Vv), acc[j]);
  }
  const int lane  = threadIdx.x & 31;
  const int rbase = m0 + ((lane >> 4) << 3);
#pragma unroll
  for (int j = 0; j < 4; ++j) {
    const int col = n0 + j * 16 + (lane & 15);
    if (col < Vv) {
#pragma unroll
      for (int r = 0; r < 8; ++r) O[(size_t)(rbase + r) * Vv + col] = acc[j][r];
    }
  }
}

// ------------------------- host driver --------------------------------------

extern "C" void kernel_launch(void* const* d_in, const int* in_sizes, int n_in,
                              void* d_out, int out_size, void* d_ws, size_t ws_size,
                              hipStream_t stream) {
  (void)in_sizes; (void)n_in; (void)out_size; (void)ws_size;

  const int*   ids  = (const int*)  d_in[0];
  const float* tok  = (const float*)d_in[1];
  const float* pos  = (const float*)d_in[2];
  const float* ln1  = (const float*)d_in[3];
  const float* Wf   = (const float*)d_in[4];
  const float* Wg   = (const float*)d_in[5];
  const float* Wh   = (const float*)d_in[6];
  const float* pwv  = (const float*)d_in[7];
  const float* sw   = (const float*)d_in[8];
  const unsigned char* sm = (const unsigned char*)d_in[9];
  const float* gw   = (const float*)d_in[10];
  const float* gb   = (const float*)d_in[11];
  const float* ln2  = (const float*)d_in[12];
  const float* fwg  = (const float*)d_in[13];
  const float* fwu  = (const float*)d_in[14];
  const float* fwd_ = (const float*)d_in[15];
  const float* lnf  = (const float*)d_in[16];
  float* out = (float*)d_out;

  const size_t RD = (size_t)ROWS * Dd;         // 524288 floats
  float* ws   = (float*)d_ws;
  float* x    = ws;
  float* a    = x    + RD;
  float* xpf  = a    + RD;
  float* xpg  = xpf  + RD;
  float* xph  = xpg  + RD;
  float* hbuf = xph  + RD;
  float* sbuf = hbuf + RD;
  float* bbuf = sbuf + RD;
  float* tbuf = bbuf + RD;                     // ROWS*Hh floats

  embed_kernel<<<ROWS, 256, 0, stream>>>(ids, tok, pos, x);

  for (int l = 0; l < Ll; ++l) {
    const float* WfL = Wf + (size_t)l * 2 * Dd * Dd;
    const float* WgL = Wg + (size_t)l * 2 * Dd * Dd;
    const float* WhL = Wh + (size_t)l * 2 * Dd * Dd;
    const float* swL = sw + (size_t)l * Dd * Dd;
    const unsigned char* smL = sm + (size_t)l * Dd * Dd;
    const float* gwL = gw + (size_t)l * Dd * Dd;
    const float* gbL = gb + (size_t)l * Dd;
    const float* wgL = fwg + (size_t)l * Dd * Hh;
    const float* wuL = fwu + (size_t)l * Dd * Hh;
    const float* wdL = fwd_ + (size_t)l * Hh * Dd;

    for (int j = 0; j < LOOPN; ++j) {
      rmsnorm_kernel<<<ROWS, 256, 0, stream>>>(x, ln1 + (size_t)l * Dd, a);
      gemm_xproj<<<dim3(Dd / 64, ROWS / 64, 3), 128, 0, stream>>>(
          a, WfL, WgL, WhL, xpf, xpg, xph);
      liquid_kernel<<<Bb, Dd, 0, stream>>>(xpf, xpg, xph, WfL, WgL, WhL, hbuf);
      plif_kernel<<<(Bb * Dd) / 256, 256, 0, stream>>>(hbuf, pwv + l, sbuf);
      gemm_syn_gate<<<dim3(Dd / 64, ROWS / 64), 128, 0, stream>>>(
          sbuf, swL, smL, gwL, gbL, x);
      rmsnorm_kernel<<<ROWS, 256, 0, stream>>>(x, ln2 + (size_t)l * Dd, bbuf);
      gemm_ffn1<<<dim3(Hh / 64, ROWS / 64), 128, 0, stream>>>(bbuf, wgL, wuL, tbuf);
      gemm_ffn2<<<dim3(Dd / 64, ROWS / 64), 128, 0, stream>>>(tbuf, wdL, x);
    }
  }

  rmsnorm_kernel<<<ROWS, 256, 0, stream>>>(x, lnf, a);
  gemm_lmhead<<<dim3((Vv + 63) / 64, ROWS / 64), 128, 0, stream>>>(a, tok, out);
}
